// RelativePositionMultiHeadAttention_31980326486428
// MI455X (gfx1250) — compile-verified, matrix operand order checked by reading, not run on hardware
//
#include <hip/hip_runtime.h>
#include <hip/hip_bf16.h>

// ---------------------------------------------------------------------------
// RelativePositionMultiHeadAttention for MI455X (gfx1250), f16 WMMA pipeline.
// B=8 T=512 M=1024 E=128 H=8, MAX_LEN=1024, offset=512.
// ---------------------------------------------------------------------------

typedef _Float16 h16;
typedef h16   v8h  __attribute__((ext_vector_type(8)));
typedef h16   v16h __attribute__((ext_vector_type(16)));
typedef float v8f  __attribute__((ext_vector_type(8)));
typedef int   i32x4 __attribute__((ext_vector_type(4)));

#define DEV __device__ __forceinline__

static constexpr int cB = 8, cT = 512, cM = 1024, cE = 128, cH = 8;
static constexpr int cBT = cB * cT;          // 4096
static constexpr int cEH = cE * cH;          // 1024
static constexpr int cMAXLEN = 1024;
static constexpr int cBHT = cB * cH * cT;    // 32768

// ---- gfx1250 async global->LDS copy (ASYNCcnt path), with safe fallback ----
#if defined(__has_builtin)
#  if __has_builtin(__builtin_amdgcn_global_load_async_to_lds_b128)
#    define HAVE_ASYNC_LDS 1
#  endif
#endif
#ifndef HAVE_ASYNC_LDS
#  define HAVE_ASYNC_LDS 0
#endif

#if HAVE_ASYNC_LDS
// copy 16 bytes global -> LDS, tracked by ASYNCcnt
// prototype (from compiler diagnostic): (v4i as(1)*, v4i as(3)*, imm off, imm cpol)
DEV void async_cp16(const void* g, void* l) {
  __builtin_amdgcn_global_load_async_to_lds_b128(
      (__attribute__((address_space(1))) i32x4*)(g),
      (__attribute__((address_space(3))) i32x4*)(l), 0, 0);
}
DEV void wait_async0() {
#  if __has_builtin(__builtin_amdgcn_s_wait_asynccnt)
  __builtin_amdgcn_s_wait_asynccnt(0);
#  else
  asm volatile("s_wait_asynccnt 0x0" ::: "memory");
#  endif
}
#else
DEV void wait_async0() {}
#endif

DEV v16h cat16(v8h lo, v8h hi) {
  return __builtin_shufflevector(lo, hi, 0,1,2,3,4,5,6,7,8,9,10,11,12,13,14,15);
}

// Load one 16x32 f16 WMMA operand fragment (A-layout: per-lane row, K-major).
// base points at element [0][0] of a row-major [rows][stride] f16 array.
// lane<16:  row=lane,    K = {0..7, 16..23}
// lane>=16: row=lane-16, K = {8..15, 24..31}
DEV v16h load_frag(const h16* base, int row0, int stride, int k0) {
  int lane = threadIdx.x & 31;
  int half = lane >> 4;
  int r    = lane & 15;
  const h16* p = base + (size_t)(row0 + r) * stride + k0 + half * 8;
  v8h lo = *(const v8h*)(p);
  v8h hi = *(const v8h*)(p + 16);
  return cat16(lo, hi);
}

DEV v8f wmma_f16(v16h a, v16h b, v8f c) {
  return __builtin_amdgcn_wmma_f32_16x16x32_f16(false, a, false, b, (short)0, c,
                                                false, false);
}

// ---------------------------------------------------------------------------
// Elementwise converts / repacks
// ---------------------------------------------------------------------------
__global__ __launch_bounds__(256) void cvt_f32_f16(const float* __restrict__ in,
                                                   h16* __restrict__ out, int n) {
  int i = blockIdx.x * 256 + threadIdx.x;
  if (i < n) out[i] = (h16)in[i];
}

// wq[m,e,h] f32 -> wt[(h*E+e)][m] f16   (n = M*E*H)
__global__ __launch_bounds__(256) void pack_w(const float* __restrict__ w,
                                              h16* __restrict__ wt, int n) {
  int i = blockIdx.x * 256 + threadIdx.x;
  if (i >= n) return;
  int m = i >> 10, rem = i & 1023;
  int e = rem >> 3, hh = rem & 7;
  wt[(size_t)(hh * cE + e) * cM + m] = (h16)w[i];
}

// wo[e,h,m] f32 -> wot[m][(h*E+e)] f16  (n = E*H*M)
__global__ __launch_bounds__(256) void pack_wo(const float* __restrict__ w,
                                               h16* __restrict__ wt, int n) {
  int i = blockIdx.x * 256 + threadIdx.x;
  if (i >= n) return;
  int e = i >> 13, rem = i & 8191;
  int hh = rem >> 10, m = rem & 1023;
  wt[(size_t)m * cEH + hh * cE + e] = (h16)w[i];
}

// ---------------------------------------------------------------------------
// Generic WMMA GEMM: C[M,N] = A[M,K] * Bt[N,K]^T, fp32 accumulate.
// Macro tile 128x64, BK=64, 256 threads = 8 waves, wave tile 32x32 (2x2 WMMA).
// Double-buffered LDS: async copies for stage kk+1 overlap WMMAs of stage kk.
// N is always 1024. Epilogue modes:
//   0: out0/out1 = f16 qu/qv in [b][h][t][e] with fp32 biases bias0/bias1[e*H+h]
//   1: out0      = f16 kh/vh in [b][h][t][e]
//   2: out0      = f16 row-major [M,N]            (rel logits table)
//   3: out0      = f32 row-major [M,N]            (final output)
// ---------------------------------------------------------------------------
__global__ __launch_bounds__(256) void gemm_wmma(
    const h16* __restrict__ A, const h16* __restrict__ Bt, int K, int mode,
    void* __restrict__ out0, void* __restrict__ out1,
    const float* __restrict__ bias0, const float* __restrict__ bias1) {
  constexpr int LSA = 72;  // padded f16 row stride (64 + 8)
  __shared__ h16 Ash[2][128 * LSA];
  __shared__ h16 Bsh[2][64 * LSA];

  const int tid = threadIdx.x;
  const int wave = tid >> 5;
  const int lane = tid & 31;
  const int wm = wave & 3, wn = wave >> 2;
  const int blockM = blockIdx.y * 128;
  const int blockN = blockIdx.x * 64;

  v8f acc[2][2] = {};

  // stage A: 128x64 f16 = 1024 x 16B chunks (4/thread)
  // stage B:  64x64 f16 =  512 x 16B chunks (2/thread)
  auto stage = [&](int buf, int k0) {
#if HAVE_ASYNC_LDS
#pragma unroll
    for (int s = 0; s < 4; ++s) {
      int vv = tid + s * 256;
      int row = vv >> 3, c8 = (vv & 7) * 8;
      async_cp16(&A[(size_t)(blockM + row) * K + k0 + c8],
                 &Ash[buf][row * LSA + c8]);
    }
#pragma unroll
    for (int s = 0; s < 2; ++s) {
      int vv = tid + s * 256;
      int row = vv >> 3, c8 = (vv & 7) * 8;
      async_cp16(&Bt[(size_t)(blockN + row) * K + k0 + c8],
                 &Bsh[buf][row * LSA + c8]);
    }
#else
#pragma unroll
    for (int s = 0; s < 4; ++s) {
      int vv = tid + s * 256;
      int row = vv >> 3, c8 = (vv & 7) * 8;
      *(v8h*)&Ash[buf][row * LSA + c8] =
          *(const v8h*)&A[(size_t)(blockM + row) * K + k0 + c8];
    }
#pragma unroll
    for (int s = 0; s < 2; ++s) {
      int vv = tid + s * 256;
      int row = vv >> 3, c8 = (vv & 7) * 8;
      *(v8h*)&Bsh[buf][row * LSA + c8] =
          *(const v8h*)&Bt[(size_t)(blockN + row) * K + k0 + c8];
    }
#endif
  };

  const int kIters = K >> 6;
  stage(0, 0);
  wait_async0();
  __syncthreads();

  for (int kk = 0; kk < kIters; ++kk) {
    const int cur = kk & 1;
    // prefetch next K-stage into the other buffer; overlaps with WMMAs below
    if (kk + 1 < kIters) stage(cur ^ 1, (kk + 1) << 6);

#pragma unroll
    for (int kc = 0; kc < 2; ++kc) {
      v16h a0 = load_frag(Ash[cur], wm * 32, LSA, kc * 32);
      v16h a1 = load_frag(Ash[cur], wm * 32 + 16, LSA, kc * 32);
      v16h b0 = load_frag(Bsh[cur], wn * 32, LSA, kc * 32);
      v16h b1 = load_frag(Bsh[cur], wn * 32 + 16, LSA, kc * 32);
      acc[0][0] = wmma_f16(a0, b0, acc[0][0]);
      acc[0][1] = wmma_f16(a0, b1, acc[0][1]);
      acc[1][0] = wmma_f16(a1, b0, acc[1][0]);
      acc[1][1] = wmma_f16(a1, b1, acc[1][1]);
    }
    wait_async0();
    __syncthreads();
  }

  // epilogue
  const int hi = lane >> 4;
  const int col = lane & 15;
#pragma unroll
  for (int ti = 0; ti < 2; ++ti) {
#pragma unroll
    for (int tj = 0; tj < 2; ++tj) {
#pragma unroll
      for (int i = 0; i < 8; ++i) {
        int gm = blockM + wm * 32 + ti * 16 + i + 8 * hi;
        int gn = blockN + wn * 32 + tj * 16 + col;
        float v = acc[ti][tj][i];
        if (mode == 2) {
          ((h16*)out0)[(size_t)gm * 1024 + gn] = (h16)v;
        } else if (mode == 3) {
          ((float*)out0)[(size_t)gm * 1024 + gn] = v;
        } else {
          // gm = b*T + t ; gn = h*E + e -> [b][h][t][e]
          int b = gm >> 9, t = gm & 511;
          int h = gn >> 7, e = gn & 127;
          size_t off = ((size_t)(b * cH + h) * cT + t) * cE + e;
          if (mode == 0) {
            ((h16*)out0)[off] = (h16)(v + bias0[e * cH + h]);
            ((h16*)out1)[off] = (h16)(v + bias1[e * cH + h]);
          } else {
            ((h16*)out0)[off] = (h16)v;
          }
        }
      }
    }
  }
}

// ---------------------------------------------------------------------------
// Flash-style attention with relative-position logits.
// grid = (B*H, T/128); block = 256 (8 waves, 16 q rows each).
// S[b,h,q,k] = (qu.kh + rel[bh, q, k-q+512]) / sqrt(E); softmax over k; P.V.
// ---------------------------------------------------------------------------
__global__ __launch_bounds__(256) void attn_rel(
    const h16* __restrict__ qu, const h16* __restrict__ kh,
    const h16* __restrict__ vh, const h16* __restrict__ rel,
    const unsigned char* __restrict__ mask, h16* __restrict__ x) {
  constexpr int KT = 32;
  constexpr int LSV = 40;  // Vt row stride (KT + 8)
  constexpr int LSS = 33;  // S row stride (KT + 1) floats
  constexpr int LSP = 40;  // P row stride

  __shared__ h16 Kt[KT * cE];          // [k][e]
  __shared__ h16 Vt[cE * LSV];         // [e][k] (transposed)
  __shared__ float Sb[8][16 * LSS];
  __shared__ h16 Pb[8][16 * LSP];
  __shared__ float wfac[8][16];

  const int tid = threadIdx.x;
  const int wave = tid >> 5;
  const int lane = tid & 31;
  const int hi = lane >> 4;
  const int col = lane & 15;

  const int bh = blockIdx.x;
  const int b = bh >> 3, h = bh & 7;
  const int q0 = blockIdx.y * 128 + wave * 16;

  const h16* qub = qu + (size_t)bh * cT * cE;
  const h16* khb = kh + (size_t)bh * cT * cE;
  const h16* vhb = vh + (size_t)bh * cT * cE;
  const h16* relb = rel + (size_t)bh * cT * cMAXLEN;
  const unsigned char* maskb = mask + b * cT;

  // Q fragments resident in registers: 16 rows x 128 e, 4 K-chunks of 32
  v16h qf[4];
#pragma unroll
  for (int c = 0; c < 4; ++c) qf[c] = load_frag(qub, q0, cE, c * 32);

  v8f o[8] = {};
  float mrow = -1e30f, lrow = 0.0f;
  const float scale = 0.08838834764831845f;  // 1/sqrt(128)

  for (int kt = 0; kt < cT / KT; ++kt) {
    const int k0 = kt * KT;
    // stage K tile [KT][E] (async path) and transposed V tile [E][KT]
#pragma unroll
    for (int s = 0; s < 2; ++s) {
      int v = tid + s * 256;                 // 512 v8h chunks
      int row = v >> 4, c8 = (v & 15) * 8;
#if HAVE_ASYNC_LDS
      async_cp16(&khb[(size_t)(k0 + row) * cE + c8], &Kt[row * cE + c8]);
#else
      *(v8h*)&Kt[row * cE + c8] =
          *(const v8h*)&khb[(size_t)(k0 + row) * cE + c8];
#endif
      v8h d = *(const v8h*)&vhb[(size_t)(k0 + row) * cE + c8];
#pragma unroll
      for (int j = 0; j < 8; ++j) Vt[(c8 + j) * LSV + row] = d[j];
    }
    wait_async0();
    __syncthreads();

    // ---- S = Q K^T (+rel, mask, scale) for this wave's 16 q rows ----
#pragma unroll
    for (int n = 0; n < 2; ++n) {
      v8f s = {};
#pragma unroll
      for (int c = 0; c < 4; ++c) {
        v16h bf = load_frag(Kt, n * 16, cE, c * 32);
        s = wmma_f16(qf[c], bf, s);
      }
      const int kcol = k0 + n * 16 + col;
      const bool mok = maskb[kcol] != 0;
#pragma unroll
      for (int i = 0; i < 8; ++i) {
        int qrow = q0 + i + 8 * hi;
        float sv = s[i] + (float)relb[(size_t)qrow * cMAXLEN + (kcol - qrow) + 512];
        sv *= scale;
        if (!mok) sv = -1e30f;
        Sb[wave][(i + 8 * hi) * LSS + n * 16 + col] = sv;
      }
    }
    __syncthreads();

    // ---- online softmax row update: lane r<16 owns q-row r ----
    if (lane < 16) {
      float* srow = &Sb[wave][lane * LSS];
      float mx = srow[0];
#pragma unroll
      for (int c = 1; c < KT; ++c) mx = fmaxf(mx, srow[c]);
      float mnew = fmaxf(mrow, mx);
      float f = __expf(mrow - mnew);
      float sum = 0.0f;
#pragma unroll
      for (int c = 0; c < KT; ++c) {
        float p = __expf(srow[c] - mnew);
        sum += p;
        Pb[wave][lane * LSP + c] = (h16)p;
      }
      lrow = lrow * f + sum;
      mrow = mnew;
      wfac[wave][lane] = f;
    }
    __syncthreads();

    // ---- rescale O and accumulate P.V ----
    float fv[16];
#pragma unroll
    for (int j = 0; j < 16; ++j) fv[j] = wfac[wave][j];
    v16h pa = load_frag(&Pb[wave][0], 0, LSP, 0);
#pragma unroll
    for (int j = 0; j < 8; ++j) {
#pragma unroll
      for (int i = 0; i < 8; ++i) o[j][i] *= fv[i + 8 * hi];
      v16h vb = load_frag(Vt, j * 16, LSV, 0);
      o[j] = wmma_f16(pa, vb, o[j]);
    }
    __syncthreads();
  }

  // ---- normalize and store x[b][q][h*E+e] ----
  if (lane < 16) wfac[wave][lane] = 1.0f / lrow;
  __syncthreads();
  float inv[16];
#pragma unroll
  for (int j = 0; j < 16; ++j) inv[j] = wfac[wave][j];
#pragma unroll
  for (int j = 0; j < 8; ++j) {
    int e = j * 16 + col;
#pragma unroll
    for (int i = 0; i < 8; ++i) {
      int qrow = q0 + i + 8 * hi;
      x[((size_t)(b * cT + qrow)) * cEH + h * cE + e] =
          (h16)(o[j][i] * inv[i + 8 * hi]);
    }
  }
}

// ---------------------------------------------------------------------------
// Host launcher
// ---------------------------------------------------------------------------
extern "C" void kernel_launch(void* const* d_in, const int* in_sizes, int n_in,
                              void* d_out, int out_size, void* d_ws,
                              size_t ws_size, hipStream_t stream) {
  const float* q   = (const float*)d_in[0];
  const float* k   = (const float*)d_in[1];
  const float* v   = (const float*)d_in[2];
  const unsigned char* mask = (const unsigned char*)d_in[3];
  const float* wq  = (const float*)d_in[4];
  const float* wk  = (const float*)d_in[5];
  const float* wv  = (const float*)d_in[6];
  const float* bu  = (const float*)d_in[7];
  const float* bv  = (const float*)d_in[8];
  const float* wo  = (const float*)d_in[9];
  const float* posw = (const float*)d_in[10];

  size_t off = 0;
  auto alloc = [&](size_t bytes) -> void* {
    void* p = (char*)d_ws + off;
    off += (bytes + 255) & ~(size_t)255;
    return p;
  };
  h16* q16   = (h16*)alloc((size_t)cBT * cM * 2);
  h16* k16   = (h16*)alloc((size_t)cBT * cM * 2);
  h16* v16   = (h16*)alloc((size_t)cBT * cM * 2);
  h16* wtq   = (h16*)alloc((size_t)cEH * cM * 2);
  h16* wtk   = (h16*)alloc((size_t)cEH * cM * 2);
  h16* wtv   = (h16*)alloc((size_t)cEH * cM * 2);
  h16* wot   = (h16*)alloc((size_t)cM * cEH * 2);
  h16* pw16  = (h16*)alloc((size_t)cMAXLEN * cE * 2);
  h16* qu    = (h16*)alloc((size_t)cBHT * cE * 2);
  h16* qv    = (h16*)alloc((size_t)cBHT * cE * 2);
  h16* khb   = (h16*)alloc((size_t)cBHT * cE * 2);
  h16* vhb   = (h16*)alloc((size_t)cBHT * cE * 2);
  h16* relb  = (h16*)alloc((size_t)cBHT * cMAXLEN * 2);   // 64 MB
  h16* xbuf  = (h16*)alloc((size_t)cBT * cEH * 2);
  (void)ws_size; (void)n_in; (void)in_sizes; (void)out_size;

  const int nqkv = cBT * cM;       // 4194304
  const int nw   = cM * cE * cH;   // 1048576
  const int npw  = cMAXLEN * cE;   // 131072

  cvt_f32_f16<<<nqkv / 256, 256, 0, stream>>>(q, q16, nqkv);
  cvt_f32_f16<<<nqkv / 256, 256, 0, stream>>>(k, k16, nqkv);
  cvt_f32_f16<<<nqkv / 256, 256, 0, stream>>>(v, v16, nqkv);
  cvt_f32_f16<<<npw / 256, 256, 0, stream>>>(posw, pw16, npw);
  pack_w<<<nw / 256, 256, 0, stream>>>(wq, wtq, nw);
  pack_w<<<nw / 256, 256, 0, stream>>>(wk, wtk, nw);
  pack_w<<<nw / 256, 256, 0, stream>>>(wv, wtv, nw);
  pack_wo<<<nw / 256, 256, 0, stream>>>(wo, wot, nw);

  // projections: [4096,1024] x [1024,1024]
  dim3 gp(1024 / 64, cBT / 128);
  gemm_wmma<<<gp, 256, 0, stream>>>(q16, wtq, cM, 0, qu, qv, bu, bv);
  gemm_wmma<<<gp, 256, 0, stream>>>(k16, wtk, cM, 1, khb, nullptr, nullptr, nullptr);
  gemm_wmma<<<gp, 256, 0, stream>>>(v16, wtv, cM, 1, vhb, nullptr, nullptr, nullptr);

  // rel logits: [32768,128] x [1024,128]^T -> [32768,1024]
  dim3 gr(1024 / 64, cBHT / 128);
  gemm_wmma<<<gr, 256, 0, stream>>>(qv, pw16, cE, 2, relb, nullptr, nullptr, nullptr);

  // attention
  dim3 ga(cB * cH, cT / 128);
  attn_rel<<<ga, 256, 0, stream>>>(qu, khb, vhb, relb, mask, xbuf);

  // output projection: [4096,1024] x [1024,1024] -> fp32 d_out
  gemm_wmma<<<gp, 256, 0, stream>>>(xbuf, wot, cEH, 3, d_out, nullptr, nullptr, nullptr);
}
